// QuestionConditionedGraphBuilder_69389491634809
// MI455X (gfx1250) — matrix operand order repaired and checked by reading, not compiled, not run on hardware
//
#include <hip/hip_runtime.h>
#include <hip/hip_bf16.h>
#include <math.h>

// ---------------------------------------------------------------------------
// QuestionConditionedGraphBuilder for MI455X (gfx1250, wave32, WMMA).
// Strategy: f16 projection table (102.4MB -> fits 192MB L2) + fused WMMA
// edge kernel; all heavy math on v_wmma_f32_16x16x32_f16.
// ---------------------------------------------------------------------------

typedef _Float16 v16h __attribute__((ext_vector_type(16)));
typedef _Float16 h8   __attribute__((ext_vector_type(8)));
typedef float    v8f  __attribute__((ext_vector_type(8)));

#define N_NODES 50000
#define N_EDGES 400000
#define NBATCH  16
#define DH      128
#define DE      64
#define DHID    256

// ---- CDNA5 WMMA fragment builders (contiguous k order, same in A and B) --------------------
// A (16x32 f16): lane l -> row M=l%16; lane group g16=l/16:
//   elems 0-15 = K = kk + 16*g16 + 0..15
__device__ __forceinline__ v16h load_a16(const _Float16* __restrict__ row,
                                         int kk, int g16) {
  h8 lo = *(const h8*)(row + kk + 16 * g16);
  h8 hi = *(const h8*)(row + kk + 16 * g16 + 8);
  v16h a;
#pragma unroll
  for (int i = 0; i < 8; ++i) { a[i] = lo[i]; a[i + 8] = hi[i]; }
  return a;
}

// B (32x16 f16): lane l -> col N=l%16; elems e -> K = kk + 16*(l/16) + e.
// We store weights N-major (row = output column, Krow elems per row), so the
// 16 halves are contiguous: two 16B loads.
__device__ __forceinline__ v16h load_b16(const _Float16* __restrict__ colptr) {
  h8 lo = *(const h8*)(colptr);
  h8 hi = *(const h8*)(colptr + 8);
  v16h b;
#pragma unroll
  for (int i = 0; i < 8; ++i) { b[i] = lo[i]; b[i + 8] = hi[i]; }
  return b;
}

// ---------------------------------------------------------------------------
// Prep kernels
// ---------------------------------------------------------------------------

// Wt[n][k] (n<1024, k<128) = f16 of concat columns [Zi|Zj|Si|Sj] weights.
__global__ void k_prep_wt(const float* __restrict__ W1g,
                          const float* __restrict__ W1s,
                          _Float16* __restrict__ Wt) {
  int idx = blockIdx.x * 256 + threadIdx.x;  // 131072 total
  int n = idx >> 7, k = idx & 127;
  float v;
  if (n < 256)      v = W1g[k * 256 + n];
  else if (n < 512) v = W1g[(128 + k) * 256 + (n - 256)];
  else if (n < 768) v = W1s[k * 256 + (n - 512)];
  else              v = W1s[(128 + k) * 256 + (n - 768)];
  Wt[n * 128 + k] = (_Float16)v;
}

// Wte[n][k] (n<256, k<64) = f16 of W1s rows 256..319 (the e-block), N-major.
__global__ void k_prep_wte(const float* __restrict__ W1s,
                           _Float16* __restrict__ Wte) {
  int idx = blockIdx.x * 256 + threadIdx.x;  // 16384 total
  int n = idx >> 6, k = idx & 63;
  Wte[n * 64 + k] = (_Float16)W1s[(256 + k) * 256 + n];
}

// Qb[b][c] = q @ W1g[256:384] + b1g  (16x256, folds b1g; kept in f32)
__global__ void k_prep_qb(const float* __restrict__ q,
                          const float* __restrict__ W1g,
                          const float* __restrict__ b1g,
                          float* __restrict__ Qb) {
  int idx = blockIdx.x * 256 + threadIdx.x;  // 4096 total
  int b = idx >> 8, c = idx & 255;
  float s = 0.f;
#pragma unroll 4
  for (int k = 0; k < 128; ++k)
    s += q[b * 128 + k] * W1g[(256 + k) * 256 + c];
  Qb[b * 256 + c] = s + b1g[c];
}

// Per-call init of segment max (ordered-uint encoding of -inf) and sums.
__global__ void k_init(unsigned* __restrict__ nodeMaxOrd,
                       float* __restrict__ sumExp) {
  int i = blockIdx.x * 256 + threadIdx.x;
  if (i < N_NODES) {
    nodeMaxOrd[i] = 0x007FFFFFu;  // encode(-inf)
    sumExp[i] = 0.f;
  }
}

// ---------------------------------------------------------------------------
// Node projection GEMM: P[N,1024](f16) = h[N,128] @ Wt^T via WMMA.
// Block = 256 thr (8 waves), one 16-row node tile; wave w -> cols [128w,128w+128).
// ---------------------------------------------------------------------------
__global__ __launch_bounds__(256)
void k_node_gemm(const float* __restrict__ h,
                 const _Float16* __restrict__ Wt,
                 _Float16* __restrict__ P) {
  __shared__ __align__(16) _Float16 Ah[16 * 128];
  const int tid = threadIdx.x;
  const int n0 = blockIdx.x * 16;
#pragma unroll
  for (int j = 0; j < 8; ++j) {
    int idx = tid + 256 * j;
    int r = idx >> 7, c = idx & 127;
    Ah[idx] = (_Float16)h[(n0 + r) * DH + c];
  }
  __syncthreads();

  const int w = tid >> 5, l = tid & 31;
  const int g16 = l >> 4, m = l & 15;
  v8f acc[8] = {};
  const _Float16* arow = Ah + m * 128;
#pragma unroll
  for (int kk = 0; kk < 128; kk += 32) {
    v16h a = load_a16(arow, kk, g16);
#pragma unroll
    for (int t = 0; t < 8; ++t) {
      int n = w * 128 + t * 16 + m;
      v16h b = load_b16(Wt + n * 128 + kk + 16 * g16);
      acc[t] = __builtin_amdgcn_wmma_f32_16x16x32_f16(
          false, a, false, b, (short)0, acc[t], false, false);
    }
  }
  // D layout: lane l VGPR g -> row g+8*(l/16), col l%16 (+tile offset)
#pragma unroll
  for (int t = 0; t < 8; ++t) {
    int col = w * 128 + t * 16 + m;
#pragma unroll
    for (int g = 0; g < 8; ++g) {
      int node = n0 + g + 8 * g16;
      P[node * 1024 + col] = (_Float16)acc[t][g];
    }
  }
}

// ---------------------------------------------------------------------------
// Fused edge kernel: Se via WMMA (b1s folded into accumulator), gathers of
// the L2-resident f16 projection table, gate & score dots via deterministic
// LDS serial reduction, raw = sigmoid(gate)*score, segment-max via atomicMax.
// Block = 128 thr (4 waves), one 16-edge tile; wave w -> cols [64w,64w+64).
// ---------------------------------------------------------------------------
__global__ __launch_bounds__(128)
void k_edge(const float* __restrict__ eFeat,
            const int* __restrict__ edge_index,
            const int* __restrict__ edge_batch,
            const _Float16* __restrict__ P,
            const _Float16* __restrict__ Wte,
            const float* __restrict__ Qb,
            const float* __restrict__ b1s,
            const float* __restrict__ W2g, const float* __restrict__ b2g,
            const float* __restrict__ W2s, const float* __restrict__ b2s,
            float* __restrict__ raw, unsigned* __restrict__ nodeMaxOrd) {
  __shared__ __align__(16) _Float16 Ae[16 * 64];
  __shared__ int msrc[16], mdst[16], mbat[16];
  __shared__ float partS[1024], partG[1024];

  const int tid = threadIdx.x;
  const int e0 = blockIdx.x * 16;
  if (tid < 16) {
    msrc[tid] = edge_index[e0 + tid];
    mdst[tid] = edge_index[N_EDGES + e0 + tid];
    mbat[tid] = edge_batch[e0 + tid];
  }
#pragma unroll
  for (int j = 0; j < 8; ++j) {
    int idx = tid + 128 * j;  // 1024 = 16 edges x 64 feats
    int r = idx >> 6, c = idx & 63;
    Ae[idx] = (_Float16)eFeat[(e0 + r) * DE + c];
  }
  __syncthreads();

  const int w = tid >> 5, l = tid & 31;
  const int g16 = l >> 4, m = l & 15;

  // Accumulators init = b1s[col] (all 8 rows of a lane share one column).
  v8f acc[4];
#pragma unroll
  for (int t = 0; t < 4; ++t) {
    float cb = b1s[w * 64 + t * 16 + m];
#pragma unroll
    for (int g = 0; g < 8; ++g) acc[t][g] = cb;
  }
  const _Float16* arow = Ae + m * 64;
#pragma unroll
  for (int kk = 0; kk < 64; kk += 32) {
    v16h a = load_a16(arow, kk, g16);
#pragma unroll
    for (int t = 0; t < 4; ++t) {
      int n = w * 64 + t * 16 + m;
      v16h b = load_b16(Wte + n * 64 + kk + 16 * g16);
      acc[t] = __builtin_amdgcn_wmma_f32_16x16x32_f16(
          false, a, false, b, (short)0, acc[t], false, false);
    }
  }

  // Epilogue: per-lane partial dots over this lane's 4 columns, 8 rows each.
  float sP[8] = {}, gP[8] = {};
#pragma unroll
  for (int t = 0; t < 4; ++t) {
    int col = w * 64 + t * 16 + m;  // hidden index 0..255
    float w2s_c = W2s[col], w2g_c = W2g[col];
#pragma unroll
    for (int g = 0; g < 8; ++g) {
      int row = g + 8 * g16;  // edge within tile; lanes 0-15 share row -> coalesced
      int s = msrc[row], d = mdst[row], bb = mbat[row];
      const _Float16* Ps = P + s * 1024;
      const _Float16* Pd = P + d * 1024;
      float sh = acc[t][g] + (float)Ps[512 + col] + (float)Pd[768 + col];
      sh = sh > 0.f ? sh : 0.f;
      sP[g] += sh * w2s_c;
      float gh = (float)Ps[col] + (float)Pd[256 + col] + Qb[bb * 256 + col];
      gh = gh > 0.f ? gh : 0.f;
      gP[g] += gh * w2g_c;
    }
  }
#pragma unroll
  for (int g = 0; g < 8; ++g) {
    partS[w * 256 + l * 8 + g] = sP[g];
    partG[w * 256 + l * 8 + g] = gP[g];
  }
  __syncthreads();

  if (tid < 16) {  // one thread per edge row; deterministic reduction order
    int grp = tid >> 3, gi = tid & 7;
    float ss = 0.f, gg = 0.f;
    for (int w2 = 0; w2 < 4; ++w2)
#pragma unroll
      for (int j = 0; j < 16; ++j) {
        int lane = grp * 16 + j;
        ss += partS[w2 * 256 + lane * 8 + gi];
        gg += partG[w2 * 256 + lane * 8 + gi];
      }
    float score = ss + b2s[0];
    float gate = 1.f / (1.f + expf(-(gg + b2g[0])));
    float rv = gate * score;
    raw[e0 + tid] = rv;
    unsigned u = __float_as_uint(rv);
    unsigned ord = (u & 0x80000000u) ? ~u : (u | 0x80000000u);
    atomicMax(&nodeMaxOrd[mdst[tid]], ord);
  }
}

// ---------------------------------------------------------------------------
// Softmax passes
// ---------------------------------------------------------------------------
__global__ void k_exp(const float* __restrict__ raw,
                      const int* __restrict__ edge_index,
                      const unsigned* __restrict__ nodeMaxOrd,
                      float* __restrict__ expv, float* __restrict__ sumExp) {
  int e = blockIdx.x * 256 + threadIdx.x;
  if (e >= N_EDGES) return;
  int d = edge_index[N_EDGES + e];
  unsigned ord = nodeMaxOrd[d];
  unsigned u = (ord & 0x80000000u) ? (ord & 0x7FFFFFFFu) : ~ord;
  float mx = __uint_as_float(u);
  float ev = expf(raw[e] - mx);
  expv[e] = ev;
  atomicAdd(&sumExp[d], ev);
}

__global__ void k_norm(const float* __restrict__ expv,
                       const int* __restrict__ edge_index,
                       const float* __restrict__ sumExp,
                       float* __restrict__ out) {
  int e = blockIdx.x * 256 + threadIdx.x;
  if (e >= N_EDGES) return;
  int d = edge_index[N_EDGES + e];
  out[e] = expv[e] / fmaxf(sumExp[d], 1e-9f);
}

// ---------------------------------------------------------------------------
extern "C" void kernel_launch(void* const* d_in, const int* in_sizes, int n_in,
                              void* d_out, int out_size, void* d_ws,
                              size_t ws_size, hipStream_t stream) {
  const float* h          = (const float*)d_in[0];
  const float* eF         = (const float*)d_in[1];
  const float* q          = (const float*)d_in[2];
  const int*   edge_index = (const int*)d_in[3];
  const int*   edge_batch = (const int*)d_in[4];
  const float* W1g        = (const float*)d_in[5];
  const float* b1g        = (const float*)d_in[6];
  const float* W2g        = (const float*)d_in[7];
  const float* b2g        = (const float*)d_in[8];
  const float* W1s        = (const float*)d_in[9];
  const float* b1s        = (const float*)d_in[10];
  const float* W2s        = (const float*)d_in[11];
  const float* b2s        = (const float*)d_in[12];
  float* out = (float*)d_out;

  char* ws = (char*)d_ws;
  size_t off = 0;
  auto take = [&](size_t bytes) {
    char* p = ws + off;
    off = (off + bytes + 255) & ~(size_t)255;
    return p;
  };
  _Float16* P     = (_Float16*)take((size_t)N_NODES * 1024 * 2);  // 102.4 MB
  _Float16* Wt    = (_Float16*)take((size_t)1024 * 128 * 2);
  _Float16* Wte   = (_Float16*)take((size_t)256 * 64 * 2);
  float*    Qb    = (float*)take((size_t)NBATCH * 256 * 4);
  float*    raw   = (float*)take((size_t)N_EDGES * 4);
  float*    expv  = (float*)take((size_t)N_EDGES * 4);
  unsigned* nmax  = (unsigned*)take((size_t)N_NODES * 4);
  float*    sume  = (float*)take((size_t)N_NODES * 4);

  k_prep_wt <<<512, 256, 0, stream>>>(W1g, W1s, Wt);
  k_prep_wte<<<64, 256, 0, stream>>>(W1s, Wte);
  k_prep_qb <<<16, 256, 0, stream>>>(q, W1g, b1g, Qb);
  k_init    <<<(N_NODES + 255) / 256, 256, 0, stream>>>(nmax, sume);

  k_node_gemm<<<N_NODES / 16, 256, 0, stream>>>(h, Wt, P);

  k_edge<<<N_EDGES / 16, 128, 0, stream>>>(eF, edge_index, edge_batch, P, Wte,
                                           Qb, b1s, W2g, b2g, W2s, b2s, raw,
                                           nmax);

  k_exp <<<(N_EDGES + 255) / 256, 256, 0, stream>>>(raw, edge_index, nmax,
                                                    expv, sume);
  k_norm<<<(N_EDGES + 255) / 256, 256, 0, stream>>>(expv, edge_index, sume,
                                                    out);
}
